// ScaledDotProductAttention_60430189855092
// MI455X (gfx1250) — matrix operand order repaired and checked by reading, not compiled, not run on hardware
//
#include <hip/hip_runtime.h>
#include <stdint.h>
#include <stddef.h>

#define D_MODEL 1024
#define HEADS   16
#define DK      64
#define NSEQ    2048

typedef __bf16        v16bf __attribute__((ext_vector_type(16)));
typedef float         v8f   __attribute__((ext_vector_type(8)));
typedef float         v4f   __attribute__((ext_vector_type(4)));
typedef unsigned int  v4u   __attribute__((ext_vector_type(4)));

union ABf16 { v4u u[2]; v16bf v; };
union F16x  { v4f f[4]; float s[16]; };
union BF16x { unsigned short s[16]; v16bf v; };

__device__ __forceinline__ unsigned short f2bf(float f) {
  unsigned int u = __float_as_uint(f);
  u += 0x7FFFu + ((u >> 16) & 1u);          // round-to-nearest-even
  return (unsigned short)(u >> 16);
}

__device__ __forceinline__ v8f zero8() {
  v8f z = {0.f,0.f,0.f,0.f,0.f,0.f,0.f,0.f};
  return z;
}

__device__ __forceinline__ v8f wmma_bf16(v16bf a, v16bf b, v8f c) {
  // D = A(16x32 bf16) * B(32x16 bf16) + C(16x16 f32)
  return __builtin_amdgcn_wmma_f32_16x16x32_bf16(false, a, false, b, (short)0, c,
                                                 false, false);
}

// B fragment: lo 8 (K=group*8+0..7) at p, hi 8 (K=16+group*8+0..7) at p+16.
__device__ __forceinline__ v16bf load_bfrag(const unsigned short* p) {
  ABf16 b;
  b.u[0] = *(const v4u*)p;
  b.u[1] = *(const v4u*)(p + 16);
  return b.v;
}

// A fragment from bf16 row-major: lo 8 (K=group*8+0..7) at p, hi 8 (K=16+group*8+0..7) at p+16.
__device__ __forceinline__ v16bf load_afrag_bf16(const unsigned short* p) {
  ABf16 a;
  a.u[0] = *(const v4u*)p;
  a.u[1] = *(const v4u*)(p + 16);
  return a.v;
}

// A fragment from f32 row-major with in-register f32->bf16 conversion.
__device__ __forceinline__ v16bf load_afrag_f32(const float* p) {
  F16x t;
  t.f[0] = *(const v4f*)(p);
  t.f[1] = *(const v4f*)(p + 4);
  t.f[2] = *(const v4f*)(p + 16);
  t.f[3] = *(const v4f*)(p + 20);
  BF16x a;
#pragma unroll
  for (int i = 0; i < 16; ++i) a.s[i] = f2bf(t.s[i]);
  return a.v;
}

// ---------------------------------------------------------------------------
// Weight transpose + bf16 convert: Wt[n][k] = (bf16) W[k][n], 1024x1024.
// ---------------------------------------------------------------------------
__global__ __launch_bounds__(256) void wtrans(const float* __restrict__ W,
                                              unsigned short* __restrict__ Wt) {
  const int k = blockIdx.x * 256 + threadIdx.x;
  const int n = blockIdx.y;
  Wt[(size_t)n * D_MODEL + k] = f2bf(W[(size_t)k * D_MODEL + n]);
}

// ---------------------------------------------------------------------------
// WMMA GEMM: rows 4096, K = 1024, N = 1024. One wave computes a 16x64 tile.
// MODE 0: A=f32, store bf16 to [b,h,seq,64]   (Q / K projection)
// MODE 1: A=f32, store bf16 to [b,h,64,seq]   (V projection, transposed)
// MODE 2: A=bf16, store f32 row-major         (output projection)
// ---------------------------------------------------------------------------
template<int MODE>
__global__ __launch_bounds__(256) void gemm16x64(const void* __restrict__ Aarg,
                                                 const unsigned short* __restrict__ Wt,
                                                 const float* __restrict__ bias,
                                                 void* __restrict__ dstArg) {
  const int tid   = threadIdx.x;
  const int lane  = tid & 31;
  const int wv    = tid >> 5;
  const int group = lane >> 4;
  const int l16   = lane & 15;
  const int tile  = blockIdx.x * 8 + wv;   // 4096 wave tiles
  const int tN    = tile & 15;             // 16 column tiles of 64
  const int tM    = tile >> 4;             // 256 row tiles of 16
  const int rowBase = tM * 16;
  const int colBase = tN * 64;

  v8f acc[4];
  acc[0] = zero8(); acc[1] = zero8(); acc[2] = zero8(); acc[3] = zero8();

  for (int k0 = 0; k0 < D_MODEL; k0 += 32) {
    v16bf a;
    if (MODE == 2) {
      const unsigned short* A = (const unsigned short*)Aarg;
      a = load_afrag_bf16(A + (size_t)(rowBase + l16) * D_MODEL + k0 + group * 8);
    } else {
      const float* A = (const float*)Aarg;
      a = load_afrag_f32(A + (size_t)(rowBase + l16) * D_MODEL + k0 + group * 8);
    }
#pragma unroll
    for (int t = 0; t < 4; ++t) {
      const unsigned short* pb =
          Wt + (size_t)(colBase + t * 16 + l16) * D_MODEL + k0 + group * 8;
      acc[t] = wmma_bf16(a, load_bfrag(pb), acc[t]);
    }
  }

#pragma unroll
  for (int t = 0; t < 4; ++t) {
    const int col = colBase + t * 16 + l16;
    const float bv = bias[col];
#pragma unroll
    for (int v = 0; v < 8; ++v) {
      const int row  = rowBase + v + group * 8;   // C layout: M = v + 8*group
      const float val = acc[t][v] + bv;
      if (MODE == 2) {
        ((float*)dstArg)[(size_t)row * D_MODEL + col] = val;
      } else {
        const int bb = row >> 11, q = row & 2047;
        const int h  = col >> 6,  d = col & 63;
        unsigned short* dst = (unsigned short*)dstArg;
        if (MODE == 0)
          dst[(((size_t)bb * HEADS + h) * NSEQ + q) * DK + d] = f2bf(val);
        else
          dst[(((size_t)bb * HEADS + h) * DK + d) * NSEQ + q] = f2bf(val);
      }
    }
  }
}

// ---------------------------------------------------------------------------
// Fused flash attention with multiplicative weights:
//   S = (Q K^T / 8) * W ; P = softmax_online(S) ; O = P V
// One wave per (b,h, 16-query tile). 32 keys per step = one WMMA K-dim.
// ---------------------------------------------------------------------------
__global__ __launch_bounds__(256) void flash_attn(const unsigned short* __restrict__ Q,
                                                  const unsigned short* __restrict__ Kc,
                                                  const unsigned short* __restrict__ Vt,
                                                  const float* __restrict__ AW,
                                                  unsigned short* __restrict__ Out) {
  __shared__ unsigned short pst[8][16 * 32];   // per-wave P staging (bf16)

  const int tid   = threadIdx.x;
  const int lane  = tid & 31;
  const int wv    = tid >> 5;
  const int group = lane >> 4;
  const int l16   = lane & 15;
  const int tile  = blockIdx.x * 8 + wv;   // 0..4095
  const int qt    = tile & 127;            // 128 query tiles
  const int bh    = tile >> 7;             // 0..31 (b*16+h)

  const unsigned short* Qb = Q  + ((size_t)bh * NSEQ + (size_t)qt * 16) * DK;
  const unsigned short* Kb = Kc + (size_t)bh * NSEQ * DK;
  const unsigned short* Vb = Vt + (size_t)bh * DK * NSEQ;
  const float*          Wb = AW + ((size_t)bh * NSEQ + (size_t)qt * 16) * NSEQ;

  v16bf qf[2];
#pragma unroll
  for (int c = 0; c < 2; ++c)
    qf[c] = load_afrag_bf16(Qb + (size_t)l16 * DK + c * 32 + group * 8);

  float mrow[8], lrow[8];
#pragma unroll
  for (int v = 0; v < 8; ++v) { mrow[v] = -3.0e38f; lrow[v] = 0.f; }
  v8f oacc[4];
  oacc[0] = zero8(); oacc[1] = zero8(); oacc[2] = zero8(); oacc[3] = zero8();

  const float scale = 0.125f;   // 1/sqrt(64)

  for (int k0 = 0; k0 < NSEQ; k0 += 32) {
    // ---- S = Q K^T (two 16x16 key subtiles, reduce over d=64 in 2 chunks) --
    v8f S[2];
#pragma unroll
    for (int st = 0; st < 2; ++st) {
      v8f s = zero8();
#pragma unroll
      for (int c = 0; c < 2; ++c) {
        const unsigned short* pb =
            Kb + (size_t)(k0 + st * 16 + l16) * DK + c * 32 + group * 8;
        s = wmma_bf16(qf[c], load_bfrag(pb), s);
      }
      S[st] = s;
    }

    if (k0 + 32 < NSEQ)
      __builtin_prefetch(Wb + (size_t)(group * 8) * NSEQ + k0 + 32, 0, 1);

    // ---- scale * attention_weights, online softmax, stage P (bf16) in LDS --
#pragma unroll
    for (int v = 0; v < 8; ++v) {
      const int row = v + group * 8;               // C layout M index
      const float* wrow = Wb + (size_t)row * NSEQ + k0;
      float s0 = S[0][v] * scale * wrow[l16];
      float s1 = S[1][v] * scale * wrow[16 + l16];
      float mx = fmaxf(s0, s1);
#pragma unroll
      for (int off = 1; off < 16; off <<= 1)
        mx = fmaxf(mx, __shfl_xor(mx, off, 32));   // reduce within 16-lane half
      const float mnew  = fmaxf(mrow[v], mx);
      const float alpha = __expf(mrow[v] - mnew);
      mrow[v] = mnew;
      const float p0 = __expf(s0 - mnew);
      const float p1 = __expf(s1 - mnew);
      float ps = p0 + p1;
#pragma unroll
      for (int off = 1; off < 16; off <<= 1)
        ps += __shfl_xor(ps, off, 32);
      lrow[v] = lrow[v] * alpha + ps;
#pragma unroll
      for (int t = 0; t < 4; ++t) oacc[t][v] *= alpha;
      pst[wv][row * 32 + l16]      = f2bf(p0);
      pst[wv][row * 32 + 16 + l16] = f2bf(p1);
    }

    // wave-local LDS RAW: wait on DS counter (no workgroup barrier needed)
    asm volatile("s_wait_dscnt 0" ::: "memory");

    const v16bf pa = load_afrag_bf16(&pst[wv][l16 * 32 + group * 8]);

    // ---- O += P V (V stored transposed: [d][n]) ---------------------------
#pragma unroll
    for (int t = 0; t < 4; ++t) {
      const unsigned short* pb =
          Vb + (size_t)(t * 16 + l16) * NSEQ + k0 + group * 8;
      oacc[t] = wmma_bf16(pa, load_bfrag(pb), oacc[t]);
    }
  }

  // ---- normalize and store bf16 [b, q, h*64+d] ----------------------------
  const int b = bh >> 4, h = bh & 15;
#pragma unroll
  for (int v = 0; v < 8; ++v) {
    const float inv = 1.0f / lrow[v];
    const int q = qt * 16 + v + group * 8;
#pragma unroll
    for (int t = 0; t < 4; ++t) {
      const int d = t * 16 + l16;
      Out[((size_t)b * NSEQ + q) * D_MODEL + (size_t)h * DK + d] =
          f2bf(oacc[t][v] * inv);
    }
  }
}

// ---------------------------------------------------------------------------
extern "C" void kernel_launch(void* const* d_in, const int* in_sizes, int n_in,
                              void* d_out, int out_size, void* d_ws, size_t ws_size,
                              hipStream_t stream) {
  const float* queries = (const float*)d_in[0];
  const float* keys    = (const float*)d_in[1];
  const float* values  = (const float*)d_in[2];
  const float* aw      = (const float*)d_in[3];
  const float* Wq      = (const float*)d_in[4];
  const float* bq      = (const float*)d_in[5];
  const float* Wk      = (const float*)d_in[6];
  const float* bk      = (const float*)d_in[7];
  const float* Wv      = (const float*)d_in[8];
  const float* bv      = (const float*)d_in[9];
  const float* Wo      = (const float*)d_in[10];
  const float* bo      = (const float*)d_in[11];

  char* ws = (char*)d_ws;
  const size_t MB = (size_t)1 << 20;
  unsigned short* wqT = (unsigned short*)(ws +  0 * MB);  // 2 MB each
  unsigned short* wkT = (unsigned short*)(ws +  2 * MB);
  unsigned short* wvT = (unsigned short*)(ws +  4 * MB);
  unsigned short* woT = (unsigned short*)(ws +  6 * MB);
  unsigned short* Qb  = (unsigned short*)(ws +  8 * MB);  // 8 MB
  unsigned short* Kb  = (unsigned short*)(ws + 16 * MB);  // 8 MB
  unsigned short* Vtb = (unsigned short*)(ws + 24 * MB);  // 8 MB
  unsigned short* Ab  = (unsigned short*)(ws + 32 * MB);  // 8 MB  (total 40 MB)

  dim3 blk(256);
  dim3 tgrid(D_MODEL / 256, D_MODEL);
  wtrans<<<tgrid, blk, 0, stream>>>(Wq, wqT);
  wtrans<<<tgrid, blk, 0, stream>>>(Wk, wkT);
  wtrans<<<tgrid, blk, 0, stream>>>(Wv, wvT);
  wtrans<<<tgrid, blk, 0, stream>>>(Wo, woT);

  gemm16x64<0><<<512, blk, 0, stream>>>((const void*)queries, wqT, bq, (void*)Qb);
  gemm16x64<0><<<512, blk, 0, stream>>>((const void*)keys,    wkT, bk, (void*)Kb);
  gemm16x64<1><<<512, blk, 0, stream>>>((const void*)values,  wvT, bv, (void*)Vtb);

  flash_attn<<<512, blk, 0, stream>>>(Qb, Kb, Vtb, aw, Ab);

  gemm16x64<2><<<512, blk, 0, stream>>>((const void*)Ab, woT, bo, d_out);

  (void)in_sizes; (void)n_in; (void)out_size; (void)ws_size;
}
